// HebbianConvBlock_36146444763689
// MI455X (gfx1250) — matrix operand order repaired and checked by reading, not compiled, not run on hardware
//
#include <hip/hip_runtime.h>
#include <hip/hip_bf16.h>
#include <cstdint>

typedef __bf16 bf16_t;
typedef __attribute__((ext_vector_type(16))) __bf16 v16bf;
typedef __attribute__((ext_vector_type(8)))  float  v8f;
typedef int v4i __attribute__((vector_size(16)));   // matches builtin param type

#define AS1 __attribute__((address_space(1)))
#define AS3 __attribute__((address_space(3)))

#if defined(__has_builtin)
#  if __has_builtin(__builtin_amdgcn_global_load_async_to_lds_b128)
#    define USE_ASYNC_LDS 1
#  endif
#endif
#ifndef USE_ASYNC_LDS
#  define USE_ASYNC_LDS 0
#endif

namespace {
constexpr int   Bc    = 4;
constexpr int   Tc    = 4096;
constexpr int   Dc    = 768;
constexpr int   Dff   = 3072;
constexpr int   Kw    = 15;      // conv taps
constexpr int   Dm    = 64;      // memory dim
constexpr float DEC   = 0.99f;
constexpr int   Mrows = Bc * Tc; // 16384

constexpr int BM = 128, BN = 128, BK = 32;
constexpr int LDSP = BK + 8;     // padded LDS stride (80B)
}

union FragBF {
  v16bf v;
  uint4 q[2];
};

__device__ __forceinline__ void wait_async_zero() {
#if USE_ASYNC_LDS
#  if defined(__has_builtin) && __has_builtin(__builtin_amdgcn_s_wait_asynccnt)
  __builtin_amdgcn_s_wait_asynccnt(0);
#  else
  asm volatile("s_wait_asynccnt 0x0" ::: "memory");
#  endif
#endif
}

#if USE_ASYNC_LDS
__device__ __forceinline__ void async_cp_b128(const bf16_t* g, bf16_t* l, int imm_off16) {
  v4i* gg = (v4i*)g;            // bitcast in generic AS
  v4i* ll = (v4i*)l;
  if (imm_off16)
    __builtin_amdgcn_global_load_async_to_lds_b128((AS1 v4i*)gg, (AS3 v4i*)ll, 16, 0);
  else
    __builtin_amdgcn_global_load_async_to_lds_b128((AS1 v4i*)gg, (AS3 v4i*)ll, 0, 0);
}
#endif

// ---------------------------------------------------------------------------
// Tiled bf16 WMMA GEMM:  C[M,N] = A[M,K] @ W[N,K]^T (+ resid)
// A: bf16 row-major [M,K]; W: bf16 row-major [N,K] (nn.Linear layout).
// 256 threads = 8 waves; waves tiled 2(M) x 4(N); each wave: 4x2 WMMA tiles.
// Double-buffered LDS; global->LDS via async DMA (ASYNCcnt) when available.
// ---------------------------------------------------------------------------
template <typename OutT, bool RESID>
__global__ __launch_bounds__(256)
void gemm_bf16_wmma(const bf16_t* __restrict__ A, const bf16_t* __restrict__ W,
                    OutT* __restrict__ C, const float* __restrict__ R,
                    int M, int N, int K) {
  __shared__ __align__(16) bf16_t As[2][BM][LDSP];
  __shared__ __align__(16) bf16_t Bs[2][BN][LDSP];

  const int tid  = threadIdx.x;
  const int lane = tid & 31;
  const int wid  = tid >> 5;
  const int wm   = wid >> 2;            // 0..1 : 64-row slab
  const int wn   = wid & 3;             // 0..3 : 32-col slab
  const int m0   = blockIdx.y * BM;
  const int n0   = blockIdx.x * BN;

  // staging: each thread moves 16 bf16 (32B) of A and of W per K-tile
  const int  srow = tid >> 1;                 // 0..127
  const int  skh  = (tid & 1) * 16;           // 0 or 16 (elements)
  const long arow = (long)(m0 + srow) * K;
  const int  wr   = (n0 + srow < N) ? (n0 + srow) : (N - 1);
  const long wrow = (long)wr * K;

  const int lm  = lane & 15;
  const int lkh = lane >> 4;

  auto stage = [&](int k0, int buf) {
#if USE_ASYNC_LDS
    // async DMA global->LDS, 2x b128 per matrix per thread; imm offset covers
    // the +16B chunk on both the global and LDS side.
    async_cp_b128(A + arow + k0 + skh, &As[buf][srow][skh], 0);
    async_cp_b128(A + arow + k0 + skh, &As[buf][srow][skh], 1);
    async_cp_b128(W + wrow + k0 + skh, &Bs[buf][srow][skh], 0);
    async_cp_b128(W + wrow + k0 + skh, &Bs[buf][srow][skh], 1);
#else
    const uint4* ga = (const uint4*)(A + arow + k0 + skh);
    uint4 a0 = ga[0], a1 = ga[1];
    const uint4* gw = (const uint4*)(W + wrow + k0 + skh);
    uint4 b0 = gw[0], b1 = gw[1];
    uint4* sa = (uint4*)&As[buf][srow][skh];
    sa[0] = a0; sa[1] = a1;
    uint4* sb = (uint4*)&Bs[buf][srow][skh];
    sb[0] = b0; sb[1] = b1;
#endif
  };

  v8f acc[4][2];
#pragma unroll
  for (int i = 0; i < 4; ++i)
#pragma unroll
    for (int j = 0; j < 2; ++j)
#pragma unroll
      for (int e = 0; e < 8; ++e) acc[i][j][e] = 0.0f;

  // prologue: tile 0 into buffer 0
  stage(0, 0);

  int buf = 0;
  for (int k0 = 0; k0 < K; k0 += BK) {
    wait_async_zero();       // my async tiles for buf have landed
    __syncthreads();         // everyone's tiles landed; other buf free to write

    if (k0 + BK < K) {
      stage(k0 + BK, buf ^ 1);   // overlap next-tile DMA with this tile's math
      if (k0 + 2 * BK < K) {     // prefetch tile k+2 -> global_prefetch_b8
        __builtin_prefetch(A + arow + k0 + 2 * BK + skh, 0, 1);
        __builtin_prefetch(W + wrow + k0 + 2 * BK + skh, 0, 1);
      }
    }

    // A fragments: lane = (M row lm, K-half lkh); chunks at k=lkh*8 and 16+lkh*8
    FragBF afr[4];
#pragma unroll
    for (int i = 0; i < 4; ++i) {
      const bf16_t* p = &As[buf][wm * 64 + i * 16 + lm][lkh * 8];
      afr[i].q[0] = *(const uint4*)p;
      afr[i].q[1] = *(const uint4*)(p + 16);
    }
    // B fragments: lane = column lm; chunks at k=lkh*8 and 16+lkh*8
    FragBF bfr[2];
#pragma unroll
    for (int j = 0; j < 2; ++j) {
      const bf16_t* p = &Bs[buf][wn * 32 + j * 16 + lm][lkh * 8];
      bfr[j].q[0] = *(const uint4*)p;
      bfr[j].q[1] = *(const uint4*)(p + 16);
    }

#pragma unroll
    for (int i = 0; i < 4; ++i)
#pragma unroll
      for (int j = 0; j < 2; ++j)
        acc[i][j] = __builtin_amdgcn_wmma_f32_16x16x32_bf16(
            false, afr[i].v, false, bfr[j].v, (short)0, acc[i][j], false, false);

    buf ^= 1;
  }

  // store: C/D layout -> row = v + 8*(lane>=16), col = lane&15
#pragma unroll
  for (int i = 0; i < 4; ++i) {
#pragma unroll
    for (int j = 0; j < 2; ++j) {
      const int col = n0 + wn * 32 + j * 16 + lm;
      if (col < N) {
        const int rbase = m0 + wm * 64 + i * 16 + lkh * 8;
#pragma unroll
        for (int v = 0; v < 8; ++v) {
          const long off = (long)(rbase + v) * N + col;
          float val = acc[i][j][v];
          if (RESID) val += R[off];
          C[off] = (OutT)val;
        }
      }
    }
  }
}

// ---------------------------------------------------------------------------
// RMSNorm (f32 in) -> bf16 out. One block per row.
// ---------------------------------------------------------------------------
__global__ __launch_bounds__(256)
void rmsnorm_bf16_kernel(const float* __restrict__ x, const float* __restrict__ w,
                         bf16_t* __restrict__ out, int D) {
  __shared__ float red[256];
  const long row = blockIdx.x;
  const float* xr = x + row * D;
  float ss = 0.0f;
  for (int i = threadIdx.x; i < D; i += 256) { float v = xr[i]; ss += v * v; }
  red[threadIdx.x] = ss;
  __syncthreads();
  for (int s = 128; s > 0; s >>= 1) {
    if (threadIdx.x < s) red[threadIdx.x] += red[threadIdx.x + s];
    __syncthreads();
  }
  const float r = rsqrtf(red[0] / (float)D + 1e-6f);
  for (int i = threadIdx.x; i < D; i += 256)
    out[row * D + i] = (bf16_t)(xr[i] * r * w[i]);
}

// ---------------------------------------------------------------------------
// f32 -> bf16 conversion (weights)
// ---------------------------------------------------------------------------
__global__ void f32_to_bf16_kernel(const float* __restrict__ in,
                                   bf16_t* __restrict__ out, long n) {
  long i = (long)blockIdx.x * 256 + threadIdx.x;
  const long stride = (long)gridDim.x * 256;
  for (; i < n; i += stride) out[i] = (bf16_t)in[i];
}

// ---------------------------------------------------------------------------
// sigmoid(gate) * causal depthwise conv(val), gv = [B,T,2D] bf16 -> bf16 [B,T,D]
// ---------------------------------------------------------------------------
__global__ __launch_bounds__(256)
void gate_conv_kernel(const bf16_t* __restrict__ gv, const float* __restrict__ cw,
                      bf16_t* __restrict__ out) {
  const int d = blockIdx.x * 256 + threadIdx.x;   // 0..767
  const int t = blockIdx.y;
  const int b = blockIdx.z;
  const long row = (long)b * Tc + t;
  float g = (float)gv[row * (2 * Dc) + d];
  g = 1.0f / (1.0f + __expf(-g));
  float s = 0.0f;
#pragma unroll
  for (int j = 0; j < Kw; ++j) {
    const int tt = t - (Kw - 1) + j;
    if (tt >= 0)
      s += (float)gv[((long)b * Tc + tt) * (2 * Dc) + Dc + d] * cw[d * Kw + j];
  }
  out[row * Dc + d] = (bf16_t)(s * g);
}

// ---------------------------------------------------------------------------
// Hebbian decay attention as a linear recurrence over a 64x64 state:
//   S_s = DEC*S_{s-1} + k[s] (x) v[s];  reads[s] = (S_s q[s]) / (8*64)
// One block per batch: 256 thr = 64 m-rows x 4 n-groups (16 state cols each).
// ---------------------------------------------------------------------------
__global__ __launch_bounds__(256)
void hebbian_scan_kernel(const float* __restrict__ kb, const float* __restrict__ vb,
                         const float* __restrict__ qb, bf16_t* __restrict__ reads) {
  __shared__ float ksh[Dm], vsh[Dm], qsh[Dm], psum[Dm * 4];
  const int tid = threadIdx.x;
  const int m   = tid & 63;
  const int g   = tid >> 6;
  const int n0  = g * 16;
  const long bbase = (long)blockIdx.x * Tc;
  const float scale = 1.0f / (8.0f * 64.0f);   // 1/(sqrt(64)*sqrt(T))

  float S[16];
#pragma unroll
  for (int i = 0; i < 16; ++i) S[i] = 0.0f;

  for (int s = 0; s < Tc; ++s) {
    const long row = (bbase + s) * Dm;
    if (tid < 64)        ksh[tid]       = kb[row + tid];
    else if (tid < 128)  vsh[tid - 64]  = vb[row + tid - 64];
    else if (tid < 192)  qsh[tid - 128] = qb[row + tid - 128];
    __syncthreads();

    const float km = ksh[m];
    float acc = 0.0f;
#pragma unroll
    for (int i = 0; i < 16; ++i) {
      S[i] = DEC * S[i] + km * vsh[n0 + i];
      acc += S[i] * qsh[n0 + i];
    }
    psum[m * 4 + g] = acc;
    __syncthreads();

    if (tid < 64) {
      const float r = psum[tid * 4] + psum[tid * 4 + 1] +
                      psum[tid * 4 + 2] + psum[tid * 4 + 3];
      reads[row + tid] = (bf16_t)(r * scale);
    }
    __syncthreads();
  }
}

// ---------------------------------------------------------------------------
// silu(g) * u, bf16 in-place into g
// ---------------------------------------------------------------------------
__global__ void swiglu_kernel(bf16_t* __restrict__ g, const bf16_t* __restrict__ u,
                              long n) {
  long i = (long)blockIdx.x * 256 + threadIdx.x;
  const long stride = (long)gridDim.x * 256;
  for (; i < n; i += stride) {
    const float gv = (float)g[i];
    const float s  = gv / (1.0f + __expf(-gv));
    g[i] = (bf16_t)(s * (float)u[i]);
  }
}

// ---------------------------------------------------------------------------
extern "C" void kernel_launch(void* const* d_in, const int* in_sizes, int n_in,
                              void* d_out, int out_size, void* d_ws, size_t ws_size,
                              hipStream_t stream) {
  (void)in_sizes; (void)n_in; (void)out_size; (void)ws_size;

  const float* x        = (const float*)d_in[0];
  const float* ln1_w    = (const float*)d_in[1];
  const float* w_up     = (const float*)d_in[2];
  const float* conv_w   = (const float*)d_in[3];
  const float* w_down   = (const float*)d_in[4];
  const float* ln_mem_w = (const float*)d_in[5];
  const float* w_k      = (const float*)d_in[6];
  const float* w_v      = (const float*)d_in[7];
  const float* w_q      = (const float*)d_in[8];
  const float* w_memo   = (const float*)d_in[9];
  const float* ln2_w    = (const float*)d_in[10];
  const float* wg       = (const float*)d_in[11];
  const float* wu       = (const float*)d_in[12];
  const float* wo       = (const float*)d_in[13];
  float* out            = (float*)d_out;

  // ---- workspace arena ----
  char* base = (char*)d_ws;
  size_t off = 0;
  auto take = [&](size_t bytes) -> void* {
    void* p = base + off;
    off = (off + bytes + 255) & ~(size_t)255;
    return p;
  };
  float*  xcur   = (float*)take((size_t)Mrows * Dc * 4);
  bf16_t* h_bf   = (bf16_t*)take((size_t)Mrows * Dc * 2);
  bf16_t* r1     = (bf16_t*)take((size_t)Mrows * Dff * 2);  // gv_bf then gbuf
  bf16_t* r2     = (bf16_t*)take((size_t)Mrows * Dff * 2);  // mix_bf then ubuf
  float*  kbuf   = (float*)take((size_t)Mrows * Dm * 4);
  float*  vbuf   = (float*)take((size_t)Mrows * Dm * 4);
  float*  qbuf   = (float*)take((size_t)Mrows * Dm * 4);
  bf16_t* rd_bf  = (bf16_t*)take((size_t)Mrows * Dm * 2);
  bf16_t* w_up_b = (bf16_t*)take((size_t)2 * Dc * Dc * 2);
  bf16_t* w_dn_b = (bf16_t*)take((size_t)Dc * Dc * 2);
  bf16_t* w_k_b  = (bf16_t*)take((size_t)Dm * Dc * 2);
  bf16_t* w_v_b  = (bf16_t*)take((size_t)Dm * Dc * 2);
  bf16_t* w_q_b  = (bf16_t*)take((size_t)Dm * Dc * 2);
  bf16_t* w_mo_b = (bf16_t*)take((size_t)Dc * Dm * 2);
  bf16_t* wg_b   = (bf16_t*)take((size_t)Dff * Dc * 2);
  bf16_t* wu_b   = (bf16_t*)take((size_t)Dff * Dc * 2);
  bf16_t* wo_b   = (bf16_t*)take((size_t)Dc * Dff * 2);

  bf16_t* gv_bf  = r1;
  bf16_t* gbuf   = r1;
  bf16_t* mix_bf = r2;
  bf16_t* ubuf   = r2;

  // ---- convert weights to bf16 ----
  auto cvt = [&](const float* src, bf16_t* dst, long n) {
    int blocks = (int)((n + 255) / 256);
    if (blocks > 8192) blocks = 8192;
    f32_to_bf16_kernel<<<blocks, 256, 0, stream>>>(src, dst, n);
  };
  cvt(w_up,   w_up_b, (long)2 * Dc * Dc);
  cvt(w_down, w_dn_b, (long)Dc * Dc);
  cvt(w_k,    w_k_b,  (long)Dm * Dc);
  cvt(w_v,    w_v_b,  (long)Dm * Dc);
  cvt(w_q,    w_q_b,  (long)Dm * Dc);
  cvt(w_memo, w_mo_b, (long)Dc * Dm);
  cvt(wg,     wg_b,   (long)Dff * Dc);
  cvt(wu,     wu_b,   (long)Dff * Dc);
  cvt(wo,     wo_b,   (long)Dc * Dff);

  const int mt = Mrows / BM;  // 128 M-tiles

  // ---- stage 1: gated conv mixer ----
  rmsnorm_bf16_kernel<<<Mrows, 256, 0, stream>>>(x, ln1_w, h_bf, Dc);
  gemm_bf16_wmma<bf16_t, false><<<dim3((2 * Dc) / BN, mt), 256, 0, stream>>>(
      h_bf, w_up_b, gv_bf, nullptr, Mrows, 2 * Dc, Dc);
  gate_conv_kernel<<<dim3(Dc / 256, Tc, Bc), 256, 0, stream>>>(gv_bf, conv_w, mix_bf);
  gemm_bf16_wmma<float, true><<<dim3(Dc / BN, mt), 256, 0, stream>>>(
      mix_bf, w_dn_b, xcur, x, Mrows, Dc, Dc);

  // ---- stage 2: Hebbian decay attention (linear recurrence form) ----
  rmsnorm_bf16_kernel<<<Mrows, 256, 0, stream>>>(xcur, ln_mem_w, h_bf, Dc);
  gemm_bf16_wmma<float, false><<<dim3(1, mt), 256, 0, stream>>>(
      h_bf, w_k_b, kbuf, nullptr, Mrows, Dm, Dc);
  gemm_bf16_wmma<float, false><<<dim3(1, mt), 256, 0, stream>>>(
      h_bf, w_v_b, vbuf, nullptr, Mrows, Dm, Dc);
  gemm_bf16_wmma<float, false><<<dim3(1, mt), 256, 0, stream>>>(
      h_bf, w_q_b, qbuf, nullptr, Mrows, Dm, Dc);
  hebbian_scan_kernel<<<Bc, 256, 0, stream>>>(kbuf, vbuf, qbuf, rd_bf);
  gemm_bf16_wmma<float, true><<<dim3(Dc / BN, mt), 256, 0, stream>>>(
      rd_bf, w_mo_b, xcur, xcur, Mrows, Dc, Dm);

  // ---- stage 3: SwiGLU FFN ----
  rmsnorm_bf16_kernel<<<Mrows, 256, 0, stream>>>(xcur, ln2_w, h_bf, Dc);
  gemm_bf16_wmma<bf16_t, false><<<dim3(Dff / BN, mt), 256, 0, stream>>>(
      h_bf, wg_b, gbuf, nullptr, Mrows, Dff, Dc);
  gemm_bf16_wmma<bf16_t, false><<<dim3(Dff / BN, mt), 256, 0, stream>>>(
      h_bf, wu_b, ubuf, nullptr, Mrows, Dff, Dc);
  swiglu_kernel<<<8192, 256, 0, stream>>>(gbuf, ubuf, (long)Mrows * Dff);
  gemm_bf16_wmma<float, true><<<dim3(Dc / BN, mt), 256, 0, stream>>>(
      gbuf, wo_b, out, xcur, Mrows, Dc, Dff);
}
